// PairformerLayer_49744311222643
// MI455X (gfx1250) — compile-verified, matrix operand order checked by reading, not run on hardware
//
#include <hip/hip_runtime.h>
#include <hip/hip_bf16.h>

typedef __attribute__((ext_vector_type(16))) __bf16 bf16x16;
typedef __attribute__((ext_vector_type(8)))  __bf16 bf16x8;
typedef __attribute__((ext_vector_type(8)))  float  floatx8;

#define DEV static __device__ __forceinline__

#if defined(__AMDGCN__) && __has_builtin(__builtin_amdgcn_global_load_async_to_lds_b128)
#define HAVE_ASYNC 1
#else
#define HAVE_ASYNC 0
#endif

typedef int v4i __attribute__((vector_size(16)));
typedef __attribute__((address_space(1))) v4i* as1_v4i_ptr;
typedef __attribute__((address_space(3))) v4i* as3_v4i_ptr;

// 16-bit WMMA A/B fragment (ISA 7.12.2): within a lane the 16 elements are two
// contiguous K-runs of 8: K = kh*8 + [0..7] and K = 16 + kh*8 + [0..7].
DEV bf16x16 frag_bf16(const __bf16* row, int kh) {
  bf16x8 lo = *(const bf16x8*)(row + (kh << 3));        // one b128 load
  bf16x8 hi = *(const bf16x8*)(row + 16 + (kh << 3));   // one b128 load
  bf16x16 f;
#pragma unroll
  for (int i = 0; i < 8; ++i) { f[i] = lo[i]; f[8 + i] = hi[i]; }
  return f;
}

DEV bf16x16 frag_f32(const float* row, int kh) {
  float4 a0 = *(const float4*)(row + (kh << 3));
  float4 a1 = *(const float4*)(row + (kh << 3) + 4);
  float4 b0 = *(const float4*)(row + 16 + (kh << 3));
  float4 b1 = *(const float4*)(row + 16 + (kh << 3) + 4);
  bf16x16 f;
  f[0]  = (__bf16)a0.x; f[1]  = (__bf16)a0.y; f[2]  = (__bf16)a0.z; f[3]  = (__bf16)a0.w;
  f[4]  = (__bf16)a1.x; f[5]  = (__bf16)a1.y; f[6]  = (__bf16)a1.z; f[7]  = (__bf16)a1.w;
  f[8]  = (__bf16)b0.x; f[9]  = (__bf16)b0.y; f[10] = (__bf16)b0.z; f[11] = (__bf16)b0.w;
  f[12] = (__bf16)b1.x; f[13] = (__bf16)b1.y; f[14] = (__bf16)b1.z; f[15] = (__bf16)b1.w;
  return f;
}

DEV floatx8 wmma_bf16(bf16x16 a, bf16x16 b, floatx8 c) {
  return __builtin_amdgcn_wmma_f32_16x16x32_bf16(false, a, false, b, (short)0, c,
                                                 false, false);
}

DEV void wait_async0() {
#if defined(__AMDGCN__) && __has_builtin(__builtin_amdgcn_s_wait_asynccnt)
  __builtin_amdgcn_s_wait_asynccnt(0);
#elif defined(__AMDGCN__)
  asm volatile("s_wait_asynccnt 0" ::: "memory");
#endif
}

// ---------------------------------------------------------------------------
// 0) zero bf16 q/k/v buffer so D=24 -> 32 pad lanes are true zeros
// ---------------------------------------------------------------------------
__global__ void zero_ws_kernel(unsigned int* p, int n) {
  int i = blockIdx.x * blockDim.x + threadIdx.x;
  if (i < n) p[i] = 0u;
}

// ---------------------------------------------------------------------------
// 1) fused q/k/v/g projections: out = s @ W.T
//    q: +bias, *1/sqrt(D); q/k -> [h][s][32] bf16; v -> [h][d(32)][s] bf16
// ---------------------------------------------------------------------------
__global__ void __launch_bounds__(32)
proj_kernel(const float* __restrict__ s,
            const float* __restrict__ qw, const float* __restrict__ qbias,
            const float* __restrict__ kw, const float* __restrict__ vw,
            const float* __restrict__ gw,
            __bf16* __restrict__ qkv, float* __restrict__ gbuf) {
  const int lane = threadIdx.x;
  const int m0 = blockIdx.x * 16, n0 = blockIdx.y * 16;
  const int mat = blockIdx.z;
  const float* W = (mat == 0) ? qw : (mat == 1) ? kw : (mat == 2) ? vw : gw;
  const int ml = lane & 15, kh = lane >> 4;
  floatx8 acc = {};
#pragma unroll 1
  for (int kc = 0; kc < 12; ++kc) {
    bf16x16 a = frag_f32(s + (size_t)(m0 + ml) * 384 + kc * 32, kh);
    bf16x16 b = frag_f32(W + (size_t)(n0 + ml) * 384 + kc * 32, kh);  // B[k][n]=W[n][k]
    acc = wmma_bf16(a, b, acc);
  }
  const int n = n0 + ml;
  const int mb = kh * 8;
  if (mat == 3) {
#pragma unroll
    for (int r = 0; r < 8; ++r)
      gbuf[(size_t)(m0 + mb + r) * 384 + n] = 1.f / (1.f + __expf(-acc[r]));
  } else {
    const int h = n / 24, d = n - h * 24;      // hoisted: n uniform over r
#pragma unroll
    for (int r = 0; r < 8; ++r) {
      int m = m0 + mb + r;
      float v = acc[r];
      if (mat == 0) v = (v + qbias[n]) * 0.20412414523193154f;  // 1/sqrt(24)
      if (mat == 2)
        qkv[(size_t)2 * 16 * 512 * 32 + ((size_t)(h * 32 + d) * 512) + m] = (__bf16)v;
      else
        qkv[(((size_t)(mat * 16 + h) * 512) + m) * 32 + d] = (__bf16)v;
    }
  }
}

// ---------------------------------------------------------------------------
// 2) pair bias: layernorm(z[q,k,:]) @ z_w.T (+ seq_mask) -> bias[H][S][S]
//    z streamed exactly once (134 MB = the roofline floor) via async->LDS
// ---------------------------------------------------------------------------
__global__ void __launch_bounds__(32)
bias_kernel(const float* __restrict__ z, const float* __restrict__ mask,
            const float* __restrict__ lnw, const float* __restrict__ lnb,
            const float* __restrict__ zw, float* __restrict__ bias) {
  __shared__ __align__(16) float zt[16][128];
  const int lane = threadIdx.x;
  const int t = blockIdx.x;
  const int qrow = t >> 5;
  const int k0 = (t & 31) << 4;
  const float* src = z + ((size_t)qrow * 512 + k0) * 128;
#if HAVE_ASYNC
#pragma unroll
  for (int i = 0; i < 16; ++i) {
    int f = i * 32 + lane;                 // 512 16-byte chunks
    int row = f >> 5, c4 = f & 31;
    __builtin_amdgcn_global_load_async_to_lds_b128(
        (as1_v4i_ptr)(src + (size_t)row * 128 + c4 * 4),
        (as3_v4i_ptr)&zt[row][c4 * 4], 0, 0);
  }
  wait_async0();
  __syncthreads();
#else
#pragma unroll
  for (int i = 0; i < 16; ++i) {
    int f = i * 32 + lane;
    int row = f >> 5, c4 = f & 31;
    float4 p = *(const float4*)(src + (size_t)row * 128 + c4 * 4);
    *(float4*)&zt[row][c4 * 4] = p;
  }
  __syncthreads();
#endif
  const int ml = lane & 15, kh = lane >> 4;
  float sum = 0.f, sq = 0.f;
  for (int c = kh * 64; c < kh * 64 + 64; ++c) {
    float x = zt[ml][c];
    sum += x; sq += x * x;
  }
  sum += __shfl_xor(sum, 16, 32);
  sq  += __shfl_xor(sq,  16, 32);
  float mu = sum * (1.f / 128.f);
  float rs = rsqrtf(sq * (1.f / 128.f) - mu * mu + 1e-5f);
  floatx8 acc = {};
#pragma unroll 1
  for (int kc = 0; kc < 4; ++kc) {
    bf16x16 a;
#pragma unroll
    for (int half = 0; half < 2; ++half) {
      const int kb = kc * 32 + (kh << 3) + half * 16;
      const float* rp = &zt[ml][kb];
#pragma unroll
      for (int i = 0; i < 8; ++i)
        a[half * 8 + i] = (__bf16)((rp[i] - mu) * rs * lnw[kb + i] + lnb[kb + i]);
    }
    bf16x16 b = frag_f32(zw + (size_t)ml * 128 + kc * 32, kh);  // B[c][h]=z_w[h][c]
    acc = wmma_bf16(a, b, acc);
  }
  const int h = ml;        // output N = head
  const int mb = kh * 8;   // output M = key-within-tile
#pragma unroll
  for (int r = 0; r < 8; ++r) {
    int kk = k0 + mb + r;
    bias[((size_t)h * 512 + qrow) * 512 + kk] = acc[r] + mask[kk];
  }
}

// ---------------------------------------------------------------------------
// 3) attention: scores = q@k.T + bias ; softmax ; o = attn @ v
// ---------------------------------------------------------------------------
__global__ void __launch_bounds__(32)
attn_kernel(const __bf16* __restrict__ qkv, const float* __restrict__ bias,
            float* __restrict__ obuf) {
  __shared__ __align__(16) float  sc[16][512];   // 32 KB
  __shared__ __align__(16) __bf16 at[16][512];   // 16 KB
  const int lane = threadIdx.x;
  const int m0 = blockIdx.x * 16;
  const int h  = blockIdx.y;
  const int ml = lane & 15, kh = lane >> 4, mb = kh * 8;
  const __bf16* qb_ = qkv + (size_t)h * 512 * 32;
  const __bf16* kb_ = qkv + (size_t)(16 + h) * 512 * 32;
  const __bf16* vb_ = qkv + (size_t)2 * 16 * 512 * 32 + (size_t)h * 32 * 512;  // [d][s]

  bf16x16 aq = frag_bf16(qb_ + (size_t)(m0 + ml) * 32, kh);  // 1/sqrt(D) folded in
#pragma unroll 1
  for (int kt = 0; kt < 32; ++kt) {
    bf16x16 bk = frag_bf16(kb_ + (size_t)(kt * 16 + ml) * 32, kh);  // B[d][key]
    floatx8 acc = {};
    acc = wmma_bf16(aq, bk, acc);
    const float* brow = bias + ((size_t)h * 512 + m0) * 512 + kt * 16 + ml;
#pragma unroll
    for (int r = 0; r < 8; ++r) {
      int m = mb + r;
      sc[m][kt * 16 + ml] = acc[r] + brow[(size_t)m * 512];
    }
  }
  __syncthreads();
  // softmax: row ml; lane-half kh scans 256 columns; combine via shfl_xor(16)
  float mx = -1e30f;
  for (int j = kh * 256; j < kh * 256 + 256; ++j) mx = fmaxf(mx, sc[ml][j]);
  mx = fmaxf(mx, __shfl_xor(mx, 16, 32));
  float ssum = 0.f;
  for (int j = kh * 256; j < kh * 256 + 256; ++j) {
    float e = __expf(sc[ml][j] - mx);
    sc[ml][j] = e;
    ssum += e;
  }
  ssum += __shfl_xor(ssum, 16, 32);
  float inv = 1.f / ssum;
  for (int j = kh * 256; j < kh * 256 + 256; ++j)
    at[ml][j] = (__bf16)(sc[ml][j] * inv);
  __syncthreads();

#pragma unroll 1
  for (int dt = 0; dt < 2; ++dt) {
    floatx8 acc = {};
#pragma unroll 1
    for (int kc = 0; kc < 16; ++kc) {
      bf16x16 a = frag_bf16(&at[ml][kc * 32], kh);                       // ds b128
      bf16x16 b = frag_bf16(vb_ + (size_t)(dt * 16 + ml) * 512 + kc * 32, kh);
      acc = wmma_bf16(a, b, acc);
    }
    int d = dt * 16 + ml;
    if (d < 24) {
#pragma unroll
      for (int r = 0; r < 8; ++r)
        obuf[(size_t)(m0 + mb + r) * 384 + h * 24 + d] = acc[r];
    }
  }
}

// ---------------------------------------------------------------------------
// 4) out = (o * sigmoid(g)) @ o_w.T
// ---------------------------------------------------------------------------
__global__ void __launch_bounds__(32)
out_kernel(const float* __restrict__ obuf, const float* __restrict__ gbuf,
           const float* __restrict__ ow, float* __restrict__ out) {
  const int lane = threadIdx.x;
  const int m0 = blockIdx.x * 16, n0 = blockIdx.y * 16;
  const int ml = lane & 15, kh = lane >> 4, mb = kh * 8;
  const float* orow = obuf + (size_t)(m0 + ml) * 384;
  const float* grow = gbuf + (size_t)(m0 + ml) * 384;
  floatx8 acc = {};
#pragma unroll 1
  for (int kc = 0; kc < 12; ++kc) {
    bf16x16 a;
#pragma unroll
    for (int half = 0; half < 2; ++half) {
      const int kb = kc * 32 + (kh << 3) + half * 16;
      float4 o0 = *(const float4*)(orow + kb);
      float4 o1 = *(const float4*)(orow + kb + 4);
      float4 g0 = *(const float4*)(grow + kb);
      float4 g1 = *(const float4*)(grow + kb + 4);
      a[half * 8 + 0] = (__bf16)(o0.x * g0.x); a[half * 8 + 1] = (__bf16)(o0.y * g0.y);
      a[half * 8 + 2] = (__bf16)(o0.z * g0.z); a[half * 8 + 3] = (__bf16)(o0.w * g0.w);
      a[half * 8 + 4] = (__bf16)(o1.x * g1.x); a[half * 8 + 5] = (__bf16)(o1.y * g1.y);
      a[half * 8 + 6] = (__bf16)(o1.z * g1.z); a[half * 8 + 7] = (__bf16)(o1.w * g1.w);
    }
    bf16x16 b = frag_f32(ow + (size_t)(n0 + ml) * 384 + kc * 32, kh);  // B[k][n]=o_w[n][k]
    acc = wmma_bf16(a, b, acc);
  }
#pragma unroll
  for (int r = 0; r < 8; ++r)
    out[(size_t)(m0 + mb + r) * 384 + n0 + ml] = acc[r];
}

extern "C" void kernel_launch(void* const* d_in, const int* in_sizes, int n_in,
                              void* d_out, int out_size, void* d_ws, size_t ws_size,
                              hipStream_t stream) {
  const float* s    = (const float*)d_in[0];
  const float* z    = (const float*)d_in[1];
  const float* mask = (const float*)d_in[2];
  const float* qw   = (const float*)d_in[3];
  const float* qb   = (const float*)d_in[4];
  const float* kw   = (const float*)d_in[5];
  const float* vw   = (const float*)d_in[6];
  const float* gw   = (const float*)d_in[7];
  const float* ow   = (const float*)d_in[8];
  const float* lnw  = (const float*)d_in[9];
  const float* lnb  = (const float*)d_in[10];
  const float* zw   = (const float*)d_in[11];
  float* out = (float*)d_out;

  // workspace layout (~19.9 MB, all L2-resident)
  __bf16* qkv = (__bf16*)d_ws;                              // 3*16*512*32 bf16
  char*   base = (char*)d_ws;
  float*  gbuf = (float*)(base + (size_t)3 * 16 * 512 * 32 * 2);
  float*  obuf = gbuf + 512 * 384;
  float*  bias = obuf + 512 * 384;                          // 16*512*512 f32

  const int qkv_dw = (3 * 16 * 512 * 32 * 2) / 4;
  zero_ws_kernel<<<dim3((qkv_dw + 255) / 256), dim3(256), 0, stream>>>(
      (unsigned int*)d_ws, qkv_dw);
  proj_kernel<<<dim3(32, 24, 4), dim3(32), 0, stream>>>(s, qw, qb, kw, vw, gw,
                                                        qkv, gbuf);
  bias_kernel<<<dim3(16384), dim3(32), 0, stream>>>(z, mask, lnw, lnb, zw, bias);
  attn_kernel<<<dim3(32, 16), dim3(32), 0, stream>>>(qkv, bias, obuf);
  out_kernel<<<dim3(32, 24), dim3(32), 0, stream>>>(obuf, gbuf, ow, out);
}
